// wModel2_58978490909302
// MI455X (gfx1250) — compile-verified, matrix operand order checked by reading, not run on hardware
//
#include <hip/hip_runtime.h>
#include <hip/hip_bf16.h>

typedef __attribute__((ext_vector_type(16))) _Float16 v16h;
typedef __attribute__((ext_vector_type(8)))  _Float16 v8h;
typedef __attribute__((ext_vector_type(4)))  _Float16 v4h;
typedef __attribute__((ext_vector_type(8)))  float    v8f;

#define NN      101   // nodes
#define MP      112   // padded rows (7 * 16)
#define NTILES  14    // 7 m-tiles * 2 n-tiles
#define NWAVES  8
#define BLK     256

// ---------------- LDS arena layout (bytes) ----------------
// att   : f16 [112][128]  @ 0      (28672)  A buffer (Kmer pad / attention matrices)
// xkh   : f16 [112][32]   @ 28672  (7168)   xk as f16 (A for head GEMMs)
// whBT  : f16 [32][128]   @ 35840  (8192)   weight B, TRANSPOSED (W_kmer / W_gat[h])
// whOT  : f16 [32][128]   @ 44032  (8192)   Wh / Wh2, TRANSPOSED (B for attention GEMMs)
// hout  : f16 [112][32]   @ 52224  (7168)   per-head output (A for Wh2 accumulation)
// wgBT  : f16 [32][32]    @ 59392  (2048)   W_gout slice, TRANSPOSED
// sv    : f32 [112]       @ 61440
// dv    : f32 [112]       @ 61888
// mu    : f32 [32]        @ 62336
// rv    : f32 [32]        @ 62464
// red   : f32 [8]         @ 62592
// awords: u32 [404]       @ 62624  (1616)   bit-packed adjacency (4 words per row)
// total 64240.  Overlays (after GAT):
//   yf (f32 [112][32]) @ 28672  (over dead xkh+whBT)
//   xv (f32 [112][32]) @ 0      (over dead att)
#define ARENA_BYTES 64240

// Load one 16-element WMMA fragment (k-contiguous row) as two b128 LDS loads.
__device__ __forceinline__ v16h frag_load(const _Float16* row, int kb, int half)
{
  union { v16h v; v8h h[2]; } u;
  u.h[0] = *(const v8h*)(row + kb + half);
  u.h[1] = *(const v8h*)(row + kb + 16 + half);
  return u.v;
}

// One 16x16 output tile of C += A(f16, row-major MxK) * B(f16, TRANSPOSED NxK).
// Wave-uniform call sites only (EXEC must be all ones for WMMA).
__device__ __forceinline__ v8f mm_tile(const _Float16* A, int lda,
                                       const _Float16* Bt, int ldb,
                                       int m0, int n0, int nk, v8f acc)
{
  const int lane = threadIdx.x & 31;
  const int half = (lane >> 4) << 3;      // 0 or 8
  const _Float16* arow = A  + (m0 + (lane & 15)) * lda;
  const _Float16* brow = Bt + (n0 + (lane & 15)) * ldb;
  for (int kt = 0; kt < nk; ++kt) {
    const int kb = kt << 5;
    v16h av = frag_load(arow, kb, half);
    v16h bv = frag_load(brow, kb, half);
    acc = __builtin_amdgcn_wmma_f32_16x16x32_f16(false, av, false, bv,
                                                 (short)0, acc, false, false);
  }
  return acc;
}

// Store 16x16 f32 tile into row-major f16 LDS matrix; zero rows >= mValid.
__device__ __forceinline__ void store_tile_f16(v8f acc, _Float16* D, int ldd,
                                               int m0, int n0, int mValid)
{
  const int lane = threadIdx.x & 31;
  const int n    = n0 + (lane & 15);
  const int mo   = (lane >> 4) << 3;
#pragma unroll
  for (int r = 0; r < 8; ++r) {
    const int m = m0 + mo + r;
    float v = (m < mValid) ? acc[r] : 0.0f;
    D[m * ldd + n] = (_Float16)v;
  }
}

// Store 16x16 f32 tile TRANSPOSED into f16 LDS matrix Dt[n][m].
__device__ __forceinline__ void store_tile_f16T(v8f acc, _Float16* Dt, int ldd,
                                                int m0, int n0, int mValid)
{
  const int lane = threadIdx.x & 31;
  const int n    = n0 + (lane & 15);
  const int mo   = (lane >> 4) << 3;
#pragma unroll
  for (int r = 0; r < 8; ++r) {
    const int m = m0 + mo + r;
    float v = (m < mValid) ? acc[r] : 0.0f;
    Dt[n * ldd + m] = (_Float16)v;
  }
}

// Masked-softmax attention matrix into att (f16 [112][128], zero-padded),
// adjacency from LDS bitmask. LeakyReLU is monotonic, so the masked max of
// leaky(s_i + d_j) equals leaky(s_i + max masked d_j).
__device__ __forceinline__ void build_att(_Float16* att, const float* sv, const float* dv,
                                          const unsigned* awords, int tid)
{
  for (int i = tid; i < MP; i += BLK) {
    _Float16* arow = att + i * 128;
    if (i < NN) {
      const unsigned a0 = awords[i * 4 + 0], a1 = awords[i * 4 + 1],
                     a2 = awords[i * 4 + 2], a3 = awords[i * 4 + 3];
      const float si = sv[i];
      // pass 1: masked max of dv
      float mxd = -3.0e38f;
      for (int j = 0; j < NN; ++j) {
        const unsigned w = (j < 32) ? a0 : (j < 64) ? a1 : (j < 96) ? a2 : a3;
        if ((w >> (j & 31)) & 1u) mxd = fmaxf(mxd, dv[j]);
      }
      const bool any = (a0 | a1 | a2 | a3) != 0u;
      float mx = 0.f;
      if (any) { const float lg = si + mxd; mx = (lg >= 0.f) ? lg : 0.2f * lg; }
      // pass 2: exp and sum (no-neighbor row -> all logits equal -> e = 1)
      float sum = 0.f;
      for (int j = 0; j < NN; ++j) {
        const unsigned w = (j < 32) ? a0 : (j < 64) ? a1 : (j < 96) ? a2 : a3;
        float e;
        if ((w >> (j & 31)) & 1u) {
          float lg = si + dv[j];
          lg = (lg >= 0.f) ? lg : 0.2f * lg;
          e = __expf(lg - mx);
        } else {
          e = any ? 0.f : 1.f;
        }
        sum += e;
        arow[j] = (_Float16)e;
      }
      const float inv = 1.0f / sum;
      for (int j = 0; j < NN; ++j) arow[j] = (_Float16)((float)arow[j] * inv);
      for (int j = NN; j < 128; ++j) arow[j] = (_Float16)0.f;
    } else {
      for (int j = 0; j < 128; ++j) arow[j] = (_Float16)0.f;
    }
  }
}

__global__ __launch_bounds__(BLK) void gat_model_kernel(
    const float* __restrict__ Kmer, const float* __restrict__ vec,
    const float* __restrict__ shp,  const int*   __restrict__ pair,
    const float* __restrict__ W_kmer, const float* __restrict__ b_kmer,
    const float* __restrict__ W_vec,  const float* __restrict__ b_vec,
    const float* __restrict__ W_gat,  const float* __restrict__ a_src,
    const float* __restrict__ a_dst,
    const float* __restrict__ W_gout, const float* __restrict__ ao_src,
    const float* __restrict__ ao_dst,
    const float* __restrict__ gamma30, const float* __restrict__ beta30,
    const float* __restrict__ W_conv,  const float* __restrict__ b_conv,
    const float* __restrict__ W_sconv, const float* __restrict__ b_sconv,
    float* __restrict__ out)
{
  __shared__ __align__(16) unsigned char smem[ARENA_BYTES];
  _Float16* att  = (_Float16*)(smem + 0);
  _Float16* xkh  = (_Float16*)(smem + 28672);
  _Float16* whBT = (_Float16*)(smem + 35840);
  _Float16* whOT = (_Float16*)(smem + 44032);
  _Float16* hout = (_Float16*)(smem + 52224);
  _Float16* wgBT = (_Float16*)(smem + 59392);
  float* sv  = (float*)(smem + 61440);
  float* dv  = (float*)(smem + 61888);
  float* mu  = (float*)(smem + 62336);
  float* rv  = (float*)(smem + 62464);
  float* red = (float*)(smem + 62592);
  unsigned* awords = (unsigned*)(smem + 62624);
  float* yf  = (float*)(smem + 28672);   // overlay, after GAT
  float* xv  = (float*)(smem + 0);       // overlay, after GAT

  const int b    = blockIdx.x;
  const int tid  = threadIdx.x;
  const int lane = tid & 31;
  const int wave = tid >> 5;
  const int* adjbase = pair + (size_t)b * NN * NN;

  // ---- stage 0a: bit-pack adjacency, one coalesced pass (wave32 ballot) ----
  for (int ww = wave; ww < NN * 4; ww += NWAVES) {
    const int i = ww >> 2, w = ww & 3;
    const int j = (w << 5) + lane;
    bool pred = false;
    if (j < NN) pred = (adjbase[i * NN + j] > 0);
    const unsigned word = __builtin_amdgcn_ballot_w32(pred);
    if (lane == 0) awords[ww] = word;
  }
  // ---- stage 0b: load Kmer (A, float4->4xf16) + W_kmer^T (B), zero whOT ----
  for (int idx = tid; idx < MP * 32; idx += BLK) {
    const int i = idx >> 5, kq = idx & 31;      // k = 4*kq
    float4 f4 = {0.f, 0.f, 0.f, 0.f};
    if (i < NN && kq < 21)
      f4 = *(const float4*)(Kmer + (size_t)b * NN * 84 + i * 84 + (kq << 2));
    v4h hv;
    hv[0] = (_Float16)f4.x; hv[1] = (_Float16)f4.y;
    hv[2] = (_Float16)f4.z; hv[3] = (_Float16)f4.w;
    *(v4h*)(att + i * 128 + (kq << 2)) = hv;
  }
  for (int idx = tid; idx < 32 * 128; idx += BLK) {
    const int n = idx >> 7, k = idx & 127;
    float v = (k < 84 && n < 30) ? W_kmer[k * 30 + n] : 0.f;
    whBT[idx] = (_Float16)v;
    whOT[idx] = (_Float16)0.f;     // zero Wh buffer once (cols >=112 stay 0)
  }
  __syncthreads();

  // ---- stage 1: xk = Kmer @ W_kmer + b_kmer  (f16 -> xkh) ----
  for (int t = wave; t < NTILES; t += NWAVES) {
    const int m0 = (t >> 1) * 16, n0 = (t & 1) * 16;
    v8f acc = {};
    acc = mm_tile(att, 128, whBT, 128, m0, n0, 3, acc);
    const int n  = n0 + (lane & 15);
    const int mo = (lane >> 4) << 3;
    const float bias = (n < 30) ? b_kmer[n] : 0.f;
#pragma unroll
    for (int r = 0; r < 8; ++r) {
      const int m = m0 + mo + r;
      float v = (m < NN && n < 30) ? (acc[r] + bias) : 0.f;
      xkh[m * 32 + n] = (_Float16)v;
    }
  }
  __syncthreads();

  // ---- stage 2: 4 GAT heads; Wh2 accumulated in registers ----
  v8f acc2a = {};            // tile slot t = wave
  v8f acc2b = {};            // tile slot t = wave + 8 (valid if < 14)
  for (int h = 0; h < 4; ++h) {
    for (int idx = tid; idx < 32 * 32; idx += BLK) {     // W_gat[h]^T -> whBT[n][k<32]
      const int n = idx >> 5, k = idx & 31;
      float v = (k < 30 && n < 30) ? W_gat[h * 900 + k * 30 + n] : 0.f;
      whBT[n * 128 + k] = (_Float16)v;
    }
    for (int idx = tid; idx < 32 * 32; idx += BLK) {     // W_gout slice^T -> wgBT
      const int n = idx >> 5, k = idx & 31;
      float v = (k < 30 && n < 30) ? W_gout[(h * 30 + k) * 30 + n] : 0.f;
      wgBT[n * 32 + k] = (_Float16)v;
    }
    __syncthreads();
    // Wh = xk @ W_gat[h]  -> whOT (transposed: [feature][node])
    for (int t = wave; t < NTILES; t += NWAVES) {
      const int m0 = (t >> 1) * 16, n0 = (t & 1) * 16;
      v8f acc = {};
      acc = mm_tile(xkh, 32, whBT, 128, m0, n0, 1, acc);
      store_tile_f16T(acc, whOT, 128, m0, n0, NN);
    }
    __syncthreads();
    // s[i] = Wh[i]@a_src, d[i] = Wh[i]@a_dst
    if (tid < MP) {
      float ss = 0.f, dd = 0.f;
      if (tid < NN) {
        for (int f = 0; f < 30; ++f) {
          const float w = (float)whOT[f * 128 + tid];
          ss += w * a_src[h * 30 + f];
          dd += w * a_dst[h * 30 + f];
        }
      }
      sv[tid] = ss; dv[tid] = dd;
    }
    __syncthreads();
    build_att(att, sv, dv, awords, tid);
    __syncthreads();
    // out_h = att @ Wh
    for (int t = wave; t < NTILES; t += NWAVES) {
      const int m0 = (t >> 1) * 16, n0 = (t & 1) * 16;
      v8f acc = {};
      acc = mm_tile(att, 128, whOT, 128, m0, n0, 4, acc);
      store_tile_f16(acc, hout, 32, m0, n0, MP);
    }
    __syncthreads();
    // Wh2 += out_h @ W_gout[h*30:(h+1)*30, :]   (register accumulation)
    {
      const int m0 = (wave >> 1) * 16, n0 = (wave & 1) * 16;
      acc2a = mm_tile(hout, 32, wgBT, 32, m0, n0, 1, acc2a);
      if (wave + 8 < NTILES) {
        const int t = wave + 8;
        acc2b = mm_tile(hout, 32, wgBT, 32, (t >> 1) * 16, (t & 1) * 16, 1, acc2b);
      }
    }
    __syncthreads();
  }

  // ---- stage 3: output GAT layer ----
  {
    const int m0 = (wave >> 1) * 16, n0 = (wave & 1) * 16;
    store_tile_f16T(acc2a, whOT, 128, m0, n0, NN);       // Wh2 -> whOT
    if (wave + 8 < NTILES) {
      const int t = wave + 8;
      store_tile_f16T(acc2b, whOT, 128, (t >> 1) * 16, (t & 1) * 16, NN);
    }
  }
  __syncthreads();
  if (tid < MP) {
    float ss = 0.f, dd = 0.f;
    if (tid < NN) {
      for (int f = 0; f < 30; ++f) {
        const float w = (float)whOT[f * 128 + tid];
        ss += w * ao_src[f];
        dd += w * ao_dst[f];
      }
    }
    sv[tid] = ss; dv[tid] = dd;
  }
  __syncthreads();
  build_att(att, sv, dv, awords, tid);
  __syncthreads();
  // y = att2 @ Wh2  (f32 into yf; yf overlays dead xkh/whBT regions)
  for (int t = wave; t < NTILES; t += NWAVES) {
    const int m0 = (t >> 1) * 16, n0 = (t & 1) * 16;
    v8f acc = {};
    acc = mm_tile(att, 128, whOT, 128, m0, n0, 4, acc);
    const int n  = n0 + (lane & 15);
    const int mo = (lane >> 4) << 3;
#pragma unroll
    for (int r = 0; r < 8; ++r) {
      const int m = m0 + mo + r;
      yf[m * 32 + n] = acc[r];
    }
  }
  __syncthreads();

  // ---- stage 4: elu + log_softmax over features ----
  if (tid < NN) {
    float* row = yf + tid * 32;
    float mx = -3.0e38f;
    for (int f = 0; f < 30; ++f) {
      float x = row[f];
      x = (x > 0.f) ? x : (__expf(x) - 1.f);   // elu
      row[f] = x;
      mx = fmaxf(mx, x);
    }
    float s = 0.f;
    for (int f = 0; f < 30; ++f) s += __expf(row[f] - mx);
    const float lse = mx + __logf(s);
    for (int f = 0; f < 30; ++f) row[f] -= lse;
  }
  __syncthreads();

  // ---- stage 5: per-sample BatchNorm(nodes) + ReLU ----
  if (tid < 30) {
    float s = 0.f, s2 = 0.f;
    for (int i = 0; i < NN; ++i) { const float x = yf[i * 32 + tid]; s += x; s2 += x * x; }
    const float m = s / (float)NN;
    mu[tid] = m;
    rv[tid] = rsqrtf(s2 / (float)NN - m * m + 1e-5f);
  }
  __syncthreads();
  for (int idx = tid; idx < NN * 30; idx += BLK) {
    const int i = idx / 30, f = idx % 30;
    const float x = (yf[i * 32 + f] - mu[f]) * rv[f] * gamma30[f] + beta30[f];
    yf[i * 32 + f] = fmaxf(x, 0.f);
  }
  // ---- stage 6: xv (einsum) into overlay at offset 0 ----
  for (int idx = tid; idx < NN * 30; idx += BLK) {
    const int i = idx / 30, f = idx % 30;
    float s = b_vec[i];
    for (int c = 0; c < 92; ++c)
      s += vec[(size_t)b * 92 * 30 + c * 30 + f] * W_vec[c * 101 + i];
    xv[i * 32 + f] = s;
  }
  if (tid < 8) red[tid] = 0.f;
  __syncthreads();

  // ---- stage 7: conv + sconv reductions ----
  float p0 = 0.f, p1 = 0.f, p2 = 0.f, p3 = 0.f, ps = 0.f;
  for (int i = tid; i < NN; i += BLK) {
    const float* w0 = W_conv + 0 * 6060 + i * 60;
    const float* w1 = W_conv + 1 * 6060 + i * 60;
    const float* w2 = W_conv + 2 * 6060 + i * 60;
    const float* w3 = W_conv + 3 * 6060 + i * 60;
    for (int k = 0; k < 30; ++k) {
      const float xa = yf[i * 32 + k];       // xg part (cols 0..29)
      const float xb = xv[i * 32 + k];       // xv part (cols 30..59)
      p0 += xa * w0[k] + xb * w0[30 + k];
      p1 += xa * w1[k] + xb * w1[30 + k];
      p2 += xa * w2[k] + xb * w2[30 + k];
      p3 += xa * w3[k] + xb * w3[30 + k];
    }
    for (int k = 0; k < 5; ++k)
      ps += shp[(size_t)b * 505 + k * 101 + i] * W_sconv[i * 5 + k];
  }
  atomicAdd(&red[0], p0); atomicAdd(&red[1], p1);
  atomicAdd(&red[2], p2); atomicAdd(&red[3], p3);
  atomicAdd(&red[4], ps);
  __syncthreads();
  if (tid == 0) {
    const float tot = red[0] + red[1] + red[2] + red[3] + red[4]
                    + b_conv[0] + b_conv[1] + b_conv[2] + b_conv[3] + b_sconv[0];
    out[b] = 1.0f / (1.0f + __expf(-tot));
  }
}

extern "C" void kernel_launch(void* const* d_in, const int* in_sizes, int n_in,
                              void* d_out, int out_size, void* d_ws, size_t ws_size,
                              hipStream_t stream) {
  (void)n_in; (void)d_ws; (void)ws_size; (void)out_size;
  const float* Kmer    = (const float*)d_in[0];
  const float* vec     = (const float*)d_in[1];
  const float* shp     = (const float*)d_in[2];
  // d_in[3] = Y (unused by the forward pass)
  const int*   pair    = (const int*)  d_in[4];
  const float* W_kmer  = (const float*)d_in[5];
  const float* b_kmer  = (const float*)d_in[6];
  const float* W_vec   = (const float*)d_in[7];
  const float* b_vec   = (const float*)d_in[8];
  const float* W_gat   = (const float*)d_in[9];
  const float* a_src   = (const float*)d_in[10];
  const float* a_dst   = (const float*)d_in[11];
  const float* W_gout  = (const float*)d_in[12];
  const float* ao_src  = (const float*)d_in[13];
  const float* ao_dst  = (const float*)d_in[14];
  const float* gamma30 = (const float*)d_in[15];
  const float* beta30  = (const float*)d_in[16];
  // d_in[17..26]: MultiheadAttention weights -> dead code in the reference output
  const float* W_conv  = (const float*)d_in[27];
  const float* b_conv  = (const float*)d_in[28];
  const float* W_sconv = (const float*)d_in[29];
  const float* b_sconv = (const float*)d_in[30];
  // d_in[31]=epoch, d_in[32]=TrainOrPre: unused

  const int Bn = in_sizes[3];   // Y has shape [B]
  gat_model_kernel<<<Bn, BLK, 0, stream>>>(
      Kmer, vec, shp, pair, W_kmer, b_kmer, W_vec, b_vec,
      W_gat, a_src, a_dst, W_gout, ao_src, ao_dst, gamma30, beta30,
      W_conv, b_conv, W_sconv, b_sconv, (float*)d_out);
}
